// Head_7730941132963
// MI455X (gfx1250) — matrix operand order repaired and checked by reading, not compiled, not run on hardware
//
#include <hip/hip_runtime.h>

#define T_LEN 4096
#define EMB   1024
#define HD    64
#define BATCH 8

typedef __attribute__((ext_vector_type(16))) __bf16 v16bf;
typedef __attribute__((ext_vector_type(8)))  __bf16 v8bf;
typedef __attribute__((ext_vector_type(8)))  float  v8f;

static __device__ inline v8bf  ld8 (const __bf16* p) { return *(const v8bf *)p; }
static __device__ inline v16bf ld16(const __bf16* p) { return *(const v16bf*)p; }
static __device__ inline v16bf cat8(v8bf lo, v8bf hi) {
  v16bf r;
#pragma unroll
  for (int i = 0; i < 8; ++i) { r[i] = lo[i]; r[8 + i] = hi[i]; }
  return r;
}
static __device__ inline v8f vzero8() {
  v8f z = {0.f, 0.f, 0.f, 0.f, 0.f, 0.f, 0.f, 0.f};
  return z;
}

// ---------------------------------------------------------------------------
// Kernel 1: fused QKV projection.  Reads x once, emits bf16 Q (pre-scaled by
// H^-0.5, row-major), K transposed [B][64][T], V row-major [B][T][64].
// One 16-row block per workgroup; wave w owns 16-col output tile w.
// ---------------------------------------------------------------------------
__global__ __launch_bounds__(128) void qkv_proj_wmma(
    const float* __restrict__ x,  const float* __restrict__ Wk,
    const float* __restrict__ Wq, const float* __restrict__ Wv,
    __bf16* __restrict__ Qb, __bf16* __restrict__ Kt, __bf16* __restrict__ Vb)
{
  const int lane = threadIdx.x & 31;
  const int half = lane >> 4;        // 0: lanes 0-15, 1: lanes 16-31
  const int l16  = lane & 15;
  const int ct   = threadIdx.x >> 5; // wave id -> 16-col tile (0..3)
  const long r0  = (long)blockIdx.x * 16;

  const float* xrow = x + (r0 + l16) * EMB; // A: lane -> row M

  v8f cq = vzero8(), ck = vzero8(), cv = vzero8();

  for (int k = 0; k < EMB; k += 32) {
    // A operand (16x32 bf16): lane L<16 holds K {0-7,16-23}, L>=16 {8-15,24-31}
    v16bf a;
    {
      const float* p0 = xrow + k + half * 8;
      const float* p1 = xrow + k + 16 + half * 8;
#pragma unroll
      for (int i = 0; i < 8; ++i) { a[i] = (__bf16)p0[i]; a[8 + i] = (__bf16)p1[i]; }
    }
    // B operand (32x16 bf16): lane -> N column, elements = K rows in A's order
    const long wr = (long)k * HD + ct * 16 + l16;
    v16bf bq, bk, bv;
#pragma unroll
    for (int i = 0; i < 16; ++i) {
      const long kk = (i < 8) ? (half * 8 + i) : (8 + half * 8 + i);
      bq[i] = (__bf16)Wq[wr + kk * HD];
      bk[i] = (__bf16)Wk[wr + kk * HD];
      bv[i] = (__bf16)Wv[wr + kk * HD];
    }
    cq = __builtin_amdgcn_wmma_f32_16x16x32_bf16(false, a, false, bq, (short)0, cq, false, false);
    ck = __builtin_amdgcn_wmma_f32_16x16x32_bf16(false, a, false, bk, (short)0, ck, false, false);
    cv = __builtin_amdgcn_wmma_f32_16x16x32_bf16(false, a, false, bv, (short)0, cv, false, false);
  }

  // C layout: VGPR i, lanes 0-15 -> M=i, N=lane; lanes 16-31 -> M=8+i
#pragma unroll
  for (int i = 0; i < 8; ++i) {
    const long g   = r0 + i + 8 * half;   // global row = b*T + t
    const int  col = ct * 16 + l16;
    Qb[g * HD + col] = (__bf16)(cq[i] * 0.125f);   // fold in H^-0.5 = 1/8
    Vb[g * HD + col] = (__bf16)cv[i];
    const long b = g >> 12, t = g & (T_LEN - 1);
    Kt[(b * HD + col) * T_LEN + t] = (__bf16)ck[i];
  }
}

// ---------------------------------------------------------------------------
// Kernel 2: causal flash attention.  Each wave owns 16 query rows, streams
// 32-key blocks with online softmax.  8 WMMAs per key block.
// ---------------------------------------------------------------------------
__global__ __launch_bounds__(128) void causal_attn_wmma(
    const __bf16* __restrict__ Qb, const __bf16* __restrict__ Kt,
    const __bf16* __restrict__ Vb, float* __restrict__ out)
{
  __shared__ __bf16 lds_p[4][16 * 32];   // per-wave P staging (C -> A layout)

  const int lane = threadIdx.x & 31;
  const int half = lane >> 4;
  const int l16  = lane & 15;
  const int wave = threadIdx.x >> 5;
  const int b    = blockIdx.x >> 6;                 // 64 query blocks / batch
  const int q0   = (blockIdx.x & 63) * 64 + wave * 16;

  const __bf16* Qrow = Qb + ((long)b * T_LEN + q0 + l16) * HD;
  const __bf16* Ktb  = Kt + (long)b * HD * T_LEN;
  const __bf16* Vbb  = Vb + (long)b * T_LEN * HD;

  // Q A operands for head dims 0-31 and 32-63 (resident for whole loop)
  const v16bf qlo = cat8(ld8(Qrow + half * 8),      ld8(Qrow + 16 + half * 8));
  const v16bf qhi = cat8(ld8(Qrow + 32 + half * 8), ld8(Qrow + 48 + half * 8));

  float m[8], l[8];
  v8f acc[4];
#pragma unroll
  for (int i = 0; i < 8; ++i) { m[i] = -__builtin_inff(); l[i] = 0.f; }
#pragma unroll
  for (int t = 0; t < 4; ++t) acc[t] = vzero8();

  __bf16* lp = &lds_p[wave][0];
  const __bf16* ktlo = Ktb + l16;                        // dims 0-31 (lane=key)
  const __bf16* kthi = Ktb + 32L * T_LEN + l16;          // dims 32-63

  for (int j = 0; j < q0 + 16; j += 32) {
    // ---- scores: S[16x32] = Q[16x64] @ K^T[64x32], two 16-key sub-tiles ----
    v8f s[2];
#pragma unroll
    for (int sub = 0; sub < 2; ++sub) {
      const int k0 = j + sub * 16;
      v16bf blo, bhi;
#pragma unroll
      for (int i = 0; i < 16; ++i) {
        const long d = (i < 8) ? (half * 8 + i) : (8 + half * 8 + i);
        blo[i] = ktlo[d * T_LEN + k0];
        bhi[i] = kthi[d * T_LEN + k0];
      }
      v8f z = vzero8();
      z      = __builtin_amdgcn_wmma_f32_16x16x32_bf16(false, qlo, false, blo, (short)0, z, false, false);
      s[sub] = __builtin_amdgcn_wmma_f32_16x16x32_bf16(false, qhi, false, bhi, (short)0, z, false, false);
    }

    // ---- causal mask + online softmax (rows live across 16-lane halves) ----
    v8f p0, p1;
    float alpha[8];
#pragma unroll
    for (int i = 0; i < 8; ++i) {
      const int row  = q0 + i + 8 * half;
      const int key0 = j + l16;
      const int key1 = key0 + 16;
      float v0 = (key0 <= row) ? s[0][i] : -__builtin_inff();
      float v1 = (key1 <= row) ? s[1][i] : -__builtin_inff();
      float mx = fmaxf(v0, v1);
#pragma unroll
      for (int off = 8; off >= 1; off >>= 1) mx = fmaxf(mx, __shfl_xor(mx, off, 16));
      const float mn = fmaxf(m[i], mx);          // finite: key j is unmasked
      const float e0 = __expf(v0 - mn);          // exp(-inf) -> 0 handles mask
      const float e1 = __expf(v1 - mn);
      p0[i] = e0; p1[i] = e1;
      float rs = e0 + e1;
#pragma unroll
      for (int off = 8; off >= 1; off >>= 1) rs += __shfl_xor(rs, off, 16);
      const float a_ = __expf(m[i] - mn);        // exp(-inf)=0 on first block
      l[i]     = l[i] * a_ + rs;
      m[i]     = mn;
      alpha[i] = a_;
    }
#pragma unroll
    for (int t = 0; t < 4; ++t)
#pragma unroll
      for (int i = 0; i < 8; ++i) acc[t][i] *= alpha[i];

    // ---- stage P through LDS: C layout -> A layout, bf16 ----
#pragma unroll
    for (int i = 0; i < 8; ++i) {
      const int row = i + 8 * half;
      lp[row * 32 + l16]      = (__bf16)p0[i];
      lp[row * 32 + 16 + l16] = (__bf16)p1[i];
    }
    asm volatile("s_wait_dscnt 0" ::: "memory");  // LDS stores visible (per-wave, in-order)

    const __bf16* pr = lp + l16 * 32;             // A: lane -> row, K chunks
    const v16bf pa = cat8(ld8(pr + half * 8), ld8(pr + 16 + half * 8));

    // ---- O += P[16x32] @ V[32x64]: lane = dim column, elements = keys in A's order ----
#pragma unroll
    for (int t = 0; t < 4; ++t) {
      v16bf bv;
#pragma unroll
      for (int i = 0; i < 16; ++i) {
        const long kk = (i < 8) ? (half * 8 + i) : (8 + half * 8 + i);
        bv[i] = Vbb[(j + kk) * HD + t * 16 + l16];
      }
      acc[t] = __builtin_amdgcn_wmma_f32_16x16x32_bf16(false, pa, false, bv, (short)0, acc[t], false, false);
    }
  }

  // ---- epilogue: normalize and write fp32 output ----
  float* orow = out + ((long)b * T_LEN + q0) * HD;
#pragma unroll
  for (int t = 0; t < 4; ++t)
#pragma unroll
    for (int i = 0; i < 8; ++i) {
      const int row = i + 8 * half;
      orow[(long)row * HD + t * 16 + l16] = acc[t][i] / l[i];
    }
}

// ---------------------------------------------------------------------------
extern "C" void kernel_launch(void* const* d_in, const int* in_sizes, int n_in,
                              void* d_out, int out_size, void* d_ws, size_t ws_size,
                              hipStream_t stream) {
  const float* x  = (const float*)d_in[0];
  const float* Wk = (const float*)d_in[1];
  const float* Wq = (const float*)d_in[2];
  const float* Wv = (const float*)d_in[3];
  float* out = (float*)d_out;

  __bf16* Qb = (__bf16*)d_ws;                       // [B*T][64] row-major
  __bf16* Kt = Qb + (size_t)BATCH * T_LEN * HD;     // [B][64][T] transposed
  __bf16* Vb = Kt + (size_t)BATCH * T_LEN * HD;     // [B*T][64] row-major

  qkv_proj_wmma<<<(BATCH * T_LEN) / 16, 128, 0, stream>>>(x, Wk, Wq, Wv, Qb, Kt, Vb);
  causal_attn_wmma<<<BATCH * (T_LEN / 64), 128, 0, stream>>>(Qb, Kt, Vb, out);
}
